// TipAdapter_68891275428268
// MI455X (gfx1250) — matrix operand order repaired and checked by reading, not compiled, not run on hardware
//
#include <hip/hip_runtime.h>

typedef _Float16     v8h   __attribute__((ext_vector_type(8)));
typedef _Float16     v16h  __attribute__((ext_vector_type(16)));
typedef float        v8f   __attribute__((ext_vector_type(8)));
typedef float        f4v   __attribute__((ext_vector_type(4)));
typedef unsigned int u32x4 __attribute__((ext_vector_type(4)));
typedef unsigned int u32x8 __attribute__((ext_vector_type(8)));

#define DDIM    512
#define NCLS    200
#define NSPLIT  32            // N-chunks for the cache GEMM
#define MBLK    128           // rows per workgroup (8 waves x 16)
#define BROW    520           // LDS B-row stride in halves (512 data + 4-dword TDM pad)
#define BBUFH   (16 * BROW)   // halves per B buffer (incl. trailing pad)

// ---- WMMA fragment loaders (CDNA5 ISA layouts, wave32) ----
// A (16x32 f16): lane L holds row M=L%16; VGPR0-3: K=8*(L/16)+{0..7}, VGPR4-7: K=16+8*(L/16)+{0..7}
__device__ __forceinline__ v16h load_frag_a(const _Float16* row, int kbase, int hv) {
  union { v16h v; struct { v8h lo, hi; } p; } u;
  u.p.lo = *(const v8h*)(row + kbase + 8 * hv);
  u.p.hi = *(const v8h*)(row + kbase + 16 + 8 * hv);
  return u.v;
}
// B (32x16 f16): lane L holds col N=L%16; VGPR0-3: K=8*(L/16)+{0..7}, VGPR4-7: K=16+8*(L/16)+{0..7}
__device__ __forceinline__ v16h load_frag_b(const _Float16* row, int kbase, int hv) {
  union { v16h v; struct { v8h lo, hi; } p; } u;
  u.p.lo = *(const v8h*)(row + kbase + 8 * hv);
  u.p.hi = *(const v8h*)(row + kbase + 16 + 8 * hv);
  return u.v;
}

// ---- TDM: DMA one 16-row x 512-f16 key tile (contiguous 16KB) into LDS ----
// D# (CDNA5 ISA): 1 descriptor, data_size=2B, tile 8192x1,
// pad_enable: every 256 dwords (one 1KB row) insert 4 dwords -> 1040B LDS row stride.
__device__ __forceinline__ void tdm_load_tile(const _Float16* gptr, unsigned lds_off) {
  unsigned long long ga = (unsigned long long)(uintptr_t)gptr;
  u32x4 g0;
  g0[0] = 1u;                                                   // count=1, user mode
  g0[1] = lds_off;                                              // LDS dest byte address
  g0[2] = (unsigned)ga;                                         // global_addr[31:0]
  g0[3] = ((unsigned)(ga >> 32) & 0x01FFFFFFu) | 0x80000000u;   // addr[56:32] | type=2
  u32x8 g1;
  g1[0] = (1u << 16) | (1u << 20) | (7u << 22) | (3u << 25);    // 2B elems, pad 4dw/256dw
  g1[1] = (8192u & 0xFFFFu) << 16;                              // tensor_dim0 = 8192
  g1[2] = (1u << 16);                                           // tensor_dim1 = 1
  g1[3] = (8192u << 16);                                        // tile_dim0 = 8192
  g1[4] = 1u;                                                   // tile_dim1 = 1
  g1[5] = 8192u;                                                // tensor_dim0_stride
  g1[6] = 0u;
  g1[7] = 0u;
  asm volatile("tensor_load_to_lds %0, %1" :: "s"(g0), "s"(g1) : "memory");
}

// ---- Pass 1: L2-normalize f32 rows -> f16 rows (zero-fill pad rows) ----
__global__ __launch_bounds__(256) void l2norm_f16_rows(
    const float* __restrict__ in, _Float16* __restrict__ out, int rows, int padRows) {
  const int wave = threadIdx.x >> 5;
  const int lane = threadIdx.x & 31;
  const int row  = blockIdx.x * 8 + wave;
  if (row >= padRows) return;
  _Float16* orow = out + (size_t)row * DDIM;
  if (row >= rows) {
    v8h z = {};
    *(v8h*)(orow + lane * 16)     = z;
    *(v8h*)(orow + lane * 16 + 8) = z;
    return;
  }
  const float* irow = in + (size_t)row * DDIM;
  float xs[16];
  float ss = 0.0f;
#pragma unroll
  for (int j = 0; j < 4; ++j) {
    f4v v = *(const f4v*)(irow + lane * 16 + j * 4);
    xs[4*j+0] = v.x; xs[4*j+1] = v.y; xs[4*j+2] = v.z; xs[4*j+3] = v.w;
    ss += v.x*v.x + v.y*v.y + v.z*v.z + v.w*v.w;
  }
#pragma unroll
  for (int off = 16; off > 0; off >>= 1) ss += __shfl_xor(ss, off, 32);
  const float r = rsqrtf(ss);
  v8h o0, o1;
#pragma unroll
  for (int i = 0; i < 8; ++i) {
    o0[i] = (_Float16)(xs[i] * r);
    o1[i] = (_Float16)(xs[8 + i] * r);
  }
  *(v8h*)(orow + lane * 16)     = o0;
  *(v8h*)(orow + lane * 16 + 8) = o1;
}

// ---- Pass 2: out = (1-ALPHA)*LOGIT_SCALE * q @ t.T  (writes every element) ----
__global__ __launch_bounds__(256) void zshot_wmma(
    const _Float16* __restrict__ qh, const _Float16* __restrict__ th,
    float* __restrict__ out, int numMTiles) {
  const int wave = threadIdx.x >> 5;
  const int lane = threadIdx.x & 31;
  const int hv   = lane >> 4;
  const int l16  = lane & 15;
  const int nTilesC = (NCLS + 15) / 16;          // 13
  const int tileId  = blockIdx.x * 8 + wave;     // wave-uniform
  if (tileId >= numMTiles * nTilesC) return;
  const int mtile = tileId / nTilesC;
  const int ntile = tileId % nTilesC;
  const _Float16* arow = qh + (size_t)(mtile * 16 + l16) * DDIM;
  const _Float16* brow = th + (size_t)(ntile * 16 + l16) * DDIM;
  v8f c = {};
#pragma unroll
  for (int s = 0; s < 16; ++s) {
    v16h a = load_frag_a(arow, s * 32, hv);
    v16h b = load_frag_b(brow, s * 32, hv);
    c = __builtin_amdgcn_wmma_f32_16x16x32_f16(false, a, false, b,
                                               (short)0, c, false, false);
  }
  const int n = ntile * 16 + l16;
  if (n < NCLS) {
#pragma unroll
    for (int r = 0; r < 8; ++r) {
      const int m = mtile * 16 + r + 8 * hv;
      out[(size_t)m * NCLS + n] = 50.0f * c[r];  // (1-ALPHA)*LOGIT_SCALE
    }
  }
}

// ---- Pass 3: fused q@keys.T -> exp(5x) -> label scatter; TDM-staged B tiles ----
__global__ __launch_bounds__(256, 1) void tip_cache_wmma(
    const _Float16* __restrict__ qh, const _Float16* __restrict__ kh,
    const int* __restrict__ labels, float* __restrict__ part,
    int numMBlocks, int numColTiles, int B) {
  __shared__ float    cLds[MBLK * NCLS];     // 100 KB class accumulator
  __shared__ _Float16 bBuf[2][BBUFH];        // 2 x 16.6 KB TDM-staged key tiles

  const int tid  = threadIdx.x;
  const int wave = tid >> 5;
  const int lane = tid & 31;
  const int hv   = lane >> 4;
  const int l16  = lane & 15;

  for (int i = tid; i < MBLK * NCLS; i += 256) cLds[i] = 0.0f;

  const int mblock = blockIdx.x % numMBlocks;
  const int nchunk = blockIdx.x / numMBlocks;
  const int mbase  = mblock * MBLK;

  // Per-wave 16-row A tile, full K=512, resident in 128 VGPRs.
  const _Float16* arow = qh + (size_t)(mbase + wave * 16 + l16) * DDIM;
  v16h a[16];
#pragma unroll
  for (int s = 0; s < 16; ++s) a[s] = load_frag_a(arow, s * 32, hv);

  // Prime the TDM pipeline: wave 0 DMAs the first tile into buffer 0.
  if (wave == 0)
    tdm_load_tile(kh + (size_t)nchunk * 16 * DDIM,
                  (unsigned)(uintptr_t)&bBuf[0][0]);

  int cur = 0;
  for (int t = nchunk; t < numColTiles; t += NSPLIT) {
    if (wave == 0) __builtin_amdgcn_s_wait_tensorcnt(0);   // bBuf[cur] filled
    __syncthreads();   // cur visible to all; all done reading bBuf[cur^1]
    const int tn = t + NSPLIT;
    if (wave == 0 && tn < numColTiles)                     // overlap next DMA
      tdm_load_tile(kh + (size_t)tn * 16 * DDIM,
                    (unsigned)(uintptr_t)&bBuf[cur ^ 1][0]);

    const _Float16* brow = &bBuf[cur][l16 * BROW];         // padded rows: no bank clash
    v8f c = {};
#pragma unroll
    for (int s = 0; s < 16; ++s) {
      v16h b = load_frag_b(brow, s * 32, hv);
      c = __builtin_amdgcn_wmma_f32_16x16x32_f16(false, a[s], false, b,
                                                 (short)0, c, false, false);
    }
    const int lbl = labels[t * 16 + l16];
#pragma unroll
    for (int r = 0; r < 8; ++r) {
      const float aff = exp2f(c[r] * 7.2134752044448169f); // exp(5x)
      atomicAdd(&cLds[(wave * 16 + r + 8 * hv) * NCLS + lbl], aff);
    }
    cur ^= 1;
  }
  __syncthreads();
  // Flush this chunk's partial c into its private slice (no global atomics).
  float* myPart = part + (size_t)nchunk * B * NCLS + (size_t)mbase * NCLS;
  for (int i = tid; i < MBLK * NCLS; i += 256) myPart[i] = cLds[i];
}

// ---- Pass 4: out += ALPHA * sum over N-chunk partials ----
__global__ __launch_bounds__(256) void reduce_partials(
    const float* __restrict__ part, float* __restrict__ out,
    int total, int nparts) {
  for (int i = blockIdx.x * 256 + threadIdx.x; i < total; i += gridDim.x * 256) {
    float s = 0.0f;
    for (int p = 0; p < nparts; ++p) s += part[(size_t)p * total + i];
    out[i] += 0.5f * s;                                    // ALPHA
  }
}

extern "C" void kernel_launch(void* const* d_in, const int* in_sizes, int n_in,
                              void* d_out, int out_size, void* d_ws, size_t ws_size,
                              hipStream_t stream) {
  const float* img    = (const float*)d_in[0];   // [B,512]
  const float* txt    = (const float*)d_in[1];   // [200,512]
  const float* keys   = (const float*)d_in[2];   // [N,512]
  const int*   labels = (const int*)d_in[3];     // [N]
  float*       out    = (float*)d_out;           // [B,200]

  const int B    = in_sizes[0] / DDIM;           // 1024
  const int C    = in_sizes[1] / DDIM;           // 200
  const int N    = in_sizes[2] / DDIM;           // 50000 (divisible by 16)
  const int Cpad = ((C + 15) / 16) * 16;         // 208

  _Float16* qh   = (_Float16*)d_ws;
  _Float16* th   = qh + (size_t)B * DDIM;
  _Float16* kh   = th + (size_t)Cpad * DDIM;
  float*    part = (float*)(kh + (size_t)N * DDIM);   // [NSPLIT][B][NCLS]

  l2norm_f16_rows<<<(B + 7) / 8,    256, 0, stream>>>(img,  qh, B, B);
  l2norm_f16_rows<<<(Cpad + 7) / 8, 256, 0, stream>>>(txt,  th, C, Cpad);
  l2norm_f16_rows<<<(N + 7) / 8,    256, 0, stream>>>(keys, kh, N, N);

  const int numMTiles = B / 16;
  const int nTilesC   = (C + 15) / 16;
  zshot_wmma<<<(numMTiles * nTilesC + 7) / 8, 256, 0, stream>>>(qh, th, out, numMTiles);

  const int numMBlocks = B / MBLK;               // 8
  tip_cache_wmma<<<numMBlocks * NSPLIT, 256, 0, stream>>>(qh, kh, labels, part,
                                                          numMBlocks, N / 16, B);
  const int total = B * NCLS;
  reduce_partials<<<(total + 255) / 256, 256, 0, stream>>>(part, out, total, NSPLIT);
}
